// GNNStack_49804440764681
// MI455X (gfx1250) — matrix operand order repaired and checked by reading, not compiled, not run on hardware
//
#include <hip/hip_runtime.h>

#define NN 50000
#define EE 800000
#define DIM 128
#define XCD 256
#define OUTD 40
#define OUTD_PAD 48

typedef __attribute__((ext_vector_type(16))) _Float16 v16h;
typedef __attribute__((ext_vector_type(8)))  float    v8f;

// ---------------------------------------------------------------- WMMA core
__device__ __forceinline__ v8f wmma_f16(v16h a, v16h b, v8f c) {
  // v_wmma_f32_16x16x32_f16: D = A(16x32 f16) * B(32x16 f16) + C(16x16 f32)
  return __builtin_amdgcn_wmma_f32_16x16x32_f16(
      /*neg_a=*/false, a, /*neg_b=*/false, b,
      /*c_mod=*/(short)0, c, /*reuse_a=*/false, /*reuse_b=*/false);
}

// K index mapping for 16-bit A-matrix 16x32 fragment (ISA 7.12.2):
// lanes 0-15: VGPR v<4 -> K=2v..2v+1 ; lanes 16-31 same rows, K+8 ; v>=4 -> K+16
__device__ __forceinline__ int kmap(int v, int half) {
  return (v < 4) ? (8 * half + 2 * v) : (16 + 8 * half + 2 * (v - 4));
}

__device__ __forceinline__ v16h load_a_f32(const float* row, int k0, int half) {
  v16h a;
#pragma unroll
  for (int v = 0; v < 8; ++v) {
    int k = k0 + kmap(v, half);
    a[2 * v]     = (_Float16)row[k];
    a[2 * v + 1] = (_Float16)row[k + 1];
  }
  return a;
}

__device__ __forceinline__ v16h load_a_h(const _Float16* row, int k0, int half) {
  v16h a;
#pragma unroll
  for (int v = 0; v < 8; ++v) {
    int k = k0 + kmap(v, half);
    a[2 * v]     = row[k];
    a[2 * v + 1] = row[k + 1];
  }
  return a;
}

// B fragment 32x16 (KxN): lane holds column n=lane&15, K [8*half,8*half+7] then [16+8*half,16+8*half+7],
// two 8-f16 chunks from weight row (W is [out][K] row-major, we compute A @ W^T).
__device__ __forceinline__ v16h load_b(const _Float16* W, int ldw, int n0, int k0, int lane) {
  int n = lane & 15, half = lane >> 4;
  typedef __attribute__((ext_vector_type(8))) _Float16 v8h_;
  const _Float16* p = W + (size_t)(n0 + n) * ldw + k0 + 8 * half;
  v8h_ lo = *(const v8h_*)p;         // 16B aligned
  v8h_ hi = *(const v8h_*)(p + 16);
  return __builtin_shufflevector(lo, hi, 0,1,2,3,4,5,6,7,8,9,10,11,12,13,14,15);
}

// ------------------------------------------- async global->LDS weight staging
// GLOBAL_LOAD_ASYNC_TO_LDS_B128 (ASYNCcnt-tracked, no VGPR data path).
// LDS operand = low 32 bits of the generic LDS-aperture address (ISA 10.2).
__device__ __forceinline__ void async_copy_b128(unsigned lds_off, const void* gaddr) {
  asm volatile("global_load_async_to_lds_b128 %0, %1, off"
               :: "v"(lds_off), "v"(gaddr) : "memory");
}
__device__ __forceinline__ void wait_async0() {
  asm volatile("s_wait_asynccnt 0x0" ::: "memory");
}

// Cooperative: copy nHalf f16 elements (multiple of 8) into sdst using async b128 ops.
__device__ __forceinline__ void stage_weights(_Float16* sdst, const _Float16* gsrc,
                                              int nHalf, int tid, int nthreads) {
  unsigned base = (unsigned)(uintptr_t)sdst;  // LDS byte offset (low 32 bits)
  int nOps = nHalf >> 3;                      // 16B per op
  for (int i = tid; i < nOps; i += nthreads)
    async_copy_b128(base + (unsigned)i * 16u, (const char*)gsrc + (size_t)i * 16);
  wait_async0();
  __syncthreads();
}

// ---------------------------------------------------------------- utility
__global__ void zero_f32(float* p, int n) {
  int i = blockIdx.x * blockDim.x + threadIdx.x;
  int stride = gridDim.x * blockDim.x;
  for (; i < n; i += stride) p[i] = 0.0f;
}

__global__ void cast_f32_to_f16(const float* s, _Float16* d, int nsrc, int ndst) {
  int i = blockIdx.x * blockDim.x + threadIdx.x;
  int stride = gridDim.x * blockDim.x;
  for (; i < ndst; i += stride) d[i] = (i < nsrc) ? (_Float16)s[i] : (_Float16)0.0f;
}

// ------------------------------------------------- K1: edge message + scatter
// wave handles 16 edges: msg = relu(x[src] @ lin_w^T + b), atomic scatter to S[dst]
__global__ __launch_bounds__(256) void msg_scatter(
    const float* __restrict__ X, const int* __restrict__ src, const int* __restrict__ dst,
    const _Float16* __restrict__ Wh, const float* __restrict__ bias,
    float* __restrict__ S, float* __restrict__ CNT, int nEdges) {
  __shared__ _Float16 sW[DIM * DIM];  // 32 KB, shared by 8 waves
  stage_weights(sW, Wh, DIM * DIM, threadIdx.x, blockDim.x);

  int wv = (blockIdx.x * blockDim.x + threadIdx.x) >> 5;
  int lane = threadIdx.x & 31;
  int e0 = wv * 16;
  if (e0 >= nEdges) return;
  int m = lane & 15, half = lane >> 4;
  int e = e0 + m;
  // warm the index stream for upcoming tiles
  if (e + 4096 < nEdges) __builtin_prefetch(&src[e + 4096], 0, 0);
  int srow = src[e];
  int drow = dst[e];
  if (half == 0) atomicAdd(&CNT[drow], 1.0f);

  const float* xr = X + (size_t)srow * DIM;
  v16h af[4];
#pragma unroll
  for (int kk = 0; kk < 4; ++kk) af[kk] = load_a_f32(xr, kk * 32, half);

  // hoist destination row offsets (element index < 6.4M, fits u32)
  unsigned roff[8];
#pragma unroll
  for (int r = 0; r < 8; ++r) {
    int dr = __shfl(drow, r + 8 * half, 32);  // lane mm owns dst[e0+mm]
    roff[r] = (unsigned)dr * (unsigned)DIM;
  }

#pragma unroll
  for (int nt = 0; nt < 8; ++nt) {
    v8f acc = {};
#pragma unroll
    for (int kk = 0; kk < 4; ++kk)
      acc = wmma_f16(af[kk], load_b(sW, DIM, nt * 16, kk * 32, lane), acc);
    int col = nt * 16 + m;
    float bn = bias[col];
#pragma unroll
    for (int r = 0; r < 8; ++r) {
      float v = acc[r] + bn;
      v = v > 0.0f ? v : 0.0f;
      atomicAdd(S + roff[r] + (unsigned)col, v);
    }
  }
}

// ------------------------------------------- K2: aggr = S/max(cnt,1); xc(f16)
__global__ void finalize_concat(const float* __restrict__ X, const float* __restrict__ S,
                                const float* __restrict__ CNT, _Float16* __restrict__ XCh) {
  int nd = blockIdx.x;
  int t = threadIdx.x;  // 128 threads
  float c = CNT[nd];
  c = c > 1.0f ? c : 1.0f;
  float a = S[(size_t)nd * DIM + t] / c;
  float x = X[(size_t)nd * DIM + t];
  XCh[(size_t)nd * XCD + t] = (_Float16)x;
  XCh[(size_t)nd * XCD + DIM + t] = (_Float16)a;
}

// ---------------------------------------------------- gate: softmax(xc@gw^T)
__global__ void gate_kernel(const _Float16* __restrict__ XCh, const float* __restrict__ gw,
                            const float* __restrict__ gb, float* __restrict__ GATE, int nNodes) {
  int wv = (blockIdx.x * blockDim.x + threadIdx.x) >> 5;
  int lane = threadIdx.x & 31;
  if (wv >= nNodes) return;
  const _Float16* xr = XCh + (size_t)wv * XCD;
  float s[4] = {0.f, 0.f, 0.f, 0.f};
  for (int c = lane; c < XCD; c += 32) {
    float xv = (float)xr[c];
#pragma unroll
    for (int eI = 0; eI < 4; ++eI) s[eI] += xv * gw[eI * XCD + c];
  }
#pragma unroll
  for (int eI = 0; eI < 4; ++eI)
#pragma unroll
    for (int off = 16; off > 0; off >>= 1) s[eI] += __shfl_xor(s[eI], off, 32);
  if (lane == 0) {
    float g0 = s[0] + gb[0], g1 = s[1] + gb[1], g2 = s[2] + gb[2], g3 = s[3] + gb[3];
    float mx = fmaxf(fmaxf(g0, g1), fmaxf(g2, g3));
    float e0 = expf(g0 - mx), e1 = expf(g1 - mx), e2 = expf(g2 - mx), e3 = expf(g3 - mx);
    float inv = 1.0f / (e0 + e1 + e2 + e3);
    GATE[(size_t)wv * 4 + 0] = e0 * inv;
    GATE[(size_t)wv * 4 + 1] = e1 * inv;
    GATE[(size_t)wv * 4 + 2] = e2 * inv;
    GATE[(size_t)wv * 4 + 3] = e3 * inv;
  }
}

// --------------------------- K3: out = sum_e gate[e]*relu(xc @ agg_w[e]^T + b)
__global__ __launch_bounds__(256) void expert_combine(
    const _Float16* __restrict__ XCh, const _Float16* __restrict__ AW,
    const float* __restrict__ ab, const float* __restrict__ GATE,
    float* __restrict__ OUT, int nNodes) {
  int wv = (blockIdx.x * blockDim.x + threadIdx.x) >> 5;
  int lane = threadIdx.x & 31;
  int r0 = wv * 16;
  if (r0 >= nNodes) return;
  int m = lane & 15, half = lane >> 4;

  const _Float16* ar = XCh + (size_t)(r0 + m) * XCD;
  v16h af[8];
#pragma unroll
  for (int kk = 0; kk < 8; ++kk) af[kk] = load_a_h(ar, kk * 32, half);

  float gv[4][8];
#pragma unroll
  for (int eI = 0; eI < 4; ++eI) {
    float gm = GATE[(size_t)(r0 + m) * 4 + eI];
#pragma unroll
    for (int r = 0; r < 8; ++r) gv[eI][r] = __shfl(gm, r + 8 * half, 32);
  }

  for (int nt = 0; nt < 8; ++nt) {
    float o[8] = {0.f, 0.f, 0.f, 0.f, 0.f, 0.f, 0.f, 0.f};
#pragma unroll
    for (int eI = 0; eI < 4; ++eI) {
      const _Float16* We = AW + (size_t)eI * DIM * XCD;
      v8f acc = {};
#pragma unroll
      for (int kk = 0; kk < 8; ++kk)
        acc = wmma_f16(af[kk], load_b(We, XCD, nt * 16, kk * 32, lane), acc);
      float be = ab[eI * DIM + nt * 16 + m];
#pragma unroll
      for (int r = 0; r < 8; ++r) {
        float v = acc[r] + be;
        v = v > 0.0f ? v : 0.0f;
        o[r] += v * gv[eI][r];
      }
    }
#pragma unroll
    for (int r = 0; r < 8; ++r)
      OUT[(size_t)(r0 + r + 8 * half) * DIM + nt * 16 + m] = o[r];
  }
}

// -------------------------------------- K4: L2 normalize per row + relu -> H
__global__ void norm_relu(const float* __restrict__ IN, float* __restrict__ H, int nNodes) {
  int wv = (blockIdx.x * blockDim.x + threadIdx.x) >> 5;
  int lane = threadIdx.x & 31;
  if (wv >= nNodes) return;
  const float* r = IN + (size_t)wv * DIM;
  float v[4];
  float ss = 0.0f;
#pragma unroll
  for (int i = 0; i < 4; ++i) {
    v[i] = r[lane + 32 * i];
    ss += v[i] * v[i];
  }
#pragma unroll
  for (int off = 16; off > 0; off >>= 1) ss += __shfl_xor(ss, off, 32);
  float nrm = sqrtf(ss);
  nrm = nrm > 1e-12f ? nrm : 1e-12f;
  float inv = 1.0f / nrm;
  float* o = H + (size_t)wv * DIM;
#pragma unroll
  for (int i = 0; i < 4; ++i) {
    float t = v[i] * inv;
    o[lane + 32 * i] = t > 0.0f ? t : 0.0f;
  }
}

// ------------------------------------------- generic GEMM: C = A @ W^T + bias
// Weights (<= DIM*DIM f16) staged in LDS via async copies.
__global__ __launch_bounds__(256) void gemm_bias(
    const _Float16* __restrict__ A, const _Float16* __restrict__ W, int wElems,
    const float* __restrict__ bias, float* __restrict__ C,
    int ldc, int outN, int nNodes) {
  __shared__ _Float16 sW[DIM * DIM];  // 32 KB
  stage_weights(sW, W, wElems, threadIdx.x, blockDim.x);

  int wv = (blockIdx.x * blockDim.x + threadIdx.x) >> 5;
  int lane = threadIdx.x & 31;
  int r0 = wv * 16;
  if (r0 >= nNodes) return;
  int m = lane & 15, half = lane >> 4;
  const _Float16* ar = A + (size_t)(r0 + m) * DIM;
  v16h af[4];
#pragma unroll
  for (int kk = 0; kk < 4; ++kk) af[kk] = load_a_h(ar, kk * 32, half);
  int ntiles = (outN + 15) >> 4;
  for (int nt = 0; nt < ntiles; ++nt) {
    v8f acc = {};
#pragma unroll
    for (int kk = 0; kk < 4; ++kk)
      acc = wmma_f16(af[kk], load_b(sW, DIM, nt * 16, kk * 32, lane), acc);
    int col = nt * 16 + m;
    if (col < outN) {
      float bn = bias[col];
#pragma unroll
      for (int r = 0; r < 8; ++r)
        C[(size_t)(r0 + r + 8 * half) * ldc + col] = acc[r] + bn;
    }
  }
}

// -------------------------------------------------- log_softmax over 40 cols
__global__ void logsoftmax40(float* __restrict__ L, int nNodes) {
  int wv = (blockIdx.x * blockDim.x + threadIdx.x) >> 5;
  int lane = threadIdx.x & 31;
  if (wv >= nNodes) return;
  float* r = L + (size_t)wv * OUTD;
  float a0 = r[lane];                       // lane < 32 < 40 always valid
  bool has1 = (lane + 32) < OUTD;           // lanes 0..7
  float a1 = has1 ? r[lane + 32] : -3.4e38f;
  float mx = fmaxf(a0, a1);
#pragma unroll
  for (int off = 16; off > 0; off >>= 1) mx = fmaxf(mx, __shfl_xor(mx, off, 32));
  float s = expf(a0 - mx) + (has1 ? expf(a1 - mx) : 0.0f);
#pragma unroll
  for (int off = 16; off > 0; off >>= 1) s += __shfl_xor(s, off, 32);
  float lse = mx + logf(s);
  r[lane] = a0 - lse;
  if (has1) r[lane + 32] = a1 - lse;
}

// ============================================================== host driver
extern "C" void kernel_launch(void* const* d_in, const int* in_sizes, int n_in,
                              void* d_out, int out_size, void* d_ws, size_t ws_size,
                              hipStream_t stream) {
  (void)in_sizes; (void)n_in; (void)out_size; (void)ws_size;
  const float* x       = (const float*)d_in[0];
  const int*   ei      = (const int*)d_in[1];
  const float* lin_w0  = (const float*)d_in[2];
  const float* lin_b0  = (const float*)d_in[3];
  const float* agg_w0  = (const float*)d_in[4];
  const float* agg_b0  = (const float*)d_in[5];
  const float* gate_w0 = (const float*)d_in[6];
  const float* gate_b0 = (const float*)d_in[7];
  const float* lin_w1  = (const float*)d_in[8];
  const float* lin_b1  = (const float*)d_in[9];
  const float* agg_w1  = (const float*)d_in[10];
  const float* agg_b1  = (const float*)d_in[11];
  const float* gate_w1 = (const float*)d_in[12];
  const float* gate_b1 = (const float*)d_in[13];
  const float* post_w1 = (const float*)d_in[14];
  const float* post_b1 = (const float*)d_in[15];
  const float* post_w2 = (const float*)d_in[16];
  const float* post_b2 = (const float*)d_in[17];
  const int* src = ei;
  const int* dst = ei + EE;
  float* out = (float*)d_out;

  // ---- workspace carve-out (aligned 256B, ~104 MB total with reuse)
  char* ws = (char*)d_ws;
  size_t off = 0;
  auto carve = [&](size_t bytes) -> char* {
    char* p = ws + off;
    off = (off + bytes + 255) & ~(size_t)255;
    return p;
  };
  _Float16* whLin0  = (_Float16*)carve((size_t)DIM * DIM * 2);
  _Float16* whLin1  = (_Float16*)carve((size_t)DIM * DIM * 2);
  _Float16* whAgg0  = (_Float16*)carve((size_t)4 * DIM * XCD * 2);
  _Float16* whAgg1  = (_Float16*)carve((size_t)4 * DIM * XCD * 2);
  _Float16* whPost1 = (_Float16*)carve((size_t)DIM * DIM * 2);
  _Float16* whPost2 = (_Float16*)carve((size_t)OUTD_PAD * DIM * 2);
  float*    S    = (float*)carve((size_t)NN * DIM * 4);
  float*    CNT  = (float*)carve((size_t)NN * 4);
  _Float16* XCh  = (_Float16*)carve((size_t)NN * XCD * 2);
  float*    GATE = (float*)carve((size_t)NN * 4 * 4);
  float*    OUT  = (float*)carve((size_t)NN * DIM * 4);  // also reused as H2
  float*    H    = (float*)carve((size_t)NN * DIM * 4);
  _Float16* Hh   = XCh;                 // reuse: post-phase f16 staging
  _Float16* Hh2  = XCh + (size_t)NN * DIM;

  const int TB = 256;
  const int edgeWaves  = EE / 16;                 // 50000
  const int edgeBlocks = edgeWaves * 32 / TB;     // 6250
  const int tileWaves  = NN / 16;                 // 3125
  const int tileBlocks = (tileWaves * 32 + TB - 1) / TB;  // 391
  const int nodeBlocks = (NN * 32 + TB - 1) / TB; // 6250 (wave per node)

  // ---- weight conversion (deterministic, cheap)
  cast_f32_to_f16<<<256, TB, 0, stream>>>(lin_w0,  whLin0,  DIM * DIM, DIM * DIM);
  cast_f32_to_f16<<<256, TB, 0, stream>>>(lin_w1,  whLin1,  DIM * DIM, DIM * DIM);
  cast_f32_to_f16<<<512, TB, 0, stream>>>(agg_w0,  whAgg0,  4 * DIM * XCD, 4 * DIM * XCD);
  cast_f32_to_f16<<<512, TB, 0, stream>>>(agg_w1,  whAgg1,  4 * DIM * XCD, 4 * DIM * XCD);
  cast_f32_to_f16<<<256, TB, 0, stream>>>(post_w1, whPost1, DIM * DIM, DIM * DIM);
  cast_f32_to_f16<<<64,  TB, 0, stream>>>(post_w2, whPost2, OUTD * DIM, OUTD_PAD * DIM);

  // ---- two GNN layers
  for (int layer = 0; layer < 2; ++layer) {
    const float*    Xin = (layer == 0) ? x : H;
    const _Float16* Wl  = (layer == 0) ? whLin0 : whLin1;
    const float*    bl  = (layer == 0) ? lin_b0 : lin_b1;
    const _Float16* Wa  = (layer == 0) ? whAgg0 : whAgg1;
    const float*    ba  = (layer == 0) ? agg_b0 : agg_b1;
    const float*    gw  = (layer == 0) ? gate_w0 : gate_w1;
    const float*    gb  = (layer == 0) ? gate_b0 : gate_b1;

    zero_f32<<<2048, TB, 0, stream>>>(S, NN * DIM);
    zero_f32<<<64,   TB, 0, stream>>>(CNT, NN);
    msg_scatter<<<edgeBlocks, TB, 0, stream>>>(Xin, src, dst, Wl, bl, S, CNT, EE);
    finalize_concat<<<NN, DIM, 0, stream>>>(Xin, S, CNT, XCh);
    gate_kernel<<<nodeBlocks, TB, 0, stream>>>(XCh, gw, gb, GATE, NN);
    expert_combine<<<tileBlocks, TB, 0, stream>>>(XCh, Wa, ba, GATE, OUT, NN);
    norm_relu<<<nodeBlocks, TB, 0, stream>>>(OUT, H, NN);
  }

  // ---- post MLP + log_softmax
  cast_f32_to_f16<<<2048, TB, 0, stream>>>(H, Hh, NN * DIM, NN * DIM);
  gemm_bias<<<tileBlocks, TB, 0, stream>>>(Hh, whPost1, DIM * DIM, post_b1, OUT, DIM, DIM, NN);
  cast_f32_to_f16<<<2048, TB, 0, stream>>>(OUT, Hh2, NN * DIM, NN * DIM);
  gemm_bias<<<tileBlocks, TB, 0, stream>>>(Hh2, whPost2, OUTD_PAD * DIM, post_b2, out, OUTD, OUTD, NN);
  logsoftmax40<<<nodeBlocks, TB, 0, stream>>>(out, NN);
}
